// AttentionPointEncoder_61778809585774
// MI455X (gfx1250) — compile-verified, matrix operand order checked by reading, not run on hardware
//
#include <hip/hip_runtime.h>
#include <cstdint>
#include <cstddef>

typedef __attribute__((ext_vector_type(16))) _Float16 v16h;
typedef __attribute__((ext_vector_type(8)))  float    v8f;

#define BB     2
#define MM     512
#define NN     2048
#define CIN    256
#define HIDN   256
#define NHEAD  8
#define DHH    32
#define INTERN 1024
#define NLAY   4
#define NCAT   2561           // 1 + 2048 + 512
#define NCATP  2592           // padded to multiple of 32
#define LNEPS  1e-5f
#define RSCALE 0.17677669529663687f  // 1/sqrt(32)

__device__ __forceinline__ v8f wmma16(v16h a, v16h b, v8f c) {
  return __builtin_amdgcn_wmma_f32_16x16x32_f16(false, a, false, b, (short)0, c,
                                                false, false);
}

// ---------------------------------------------------------------------------
// Generic WMMA GEMM:  out[rows,Ncols] = A[rows,K](f16) @ Wt[Ncols,K](f16,K-major)
// epilogue: + bias, + resid(f32,[rows,Ncols]), optional relu; writes f32 and/or f16.
// Block = 256 threads = 8 waves; each wave: 32 rows x 64 cols (2x4 acc tiles,
// 8 WMMAs per 32-wide k-step, B fragments reused 2x, A fragments reused 4x).
// A rows are CLAMPED (not predicated) so the k-loop has no EXEC manipulation;
// out-of-range rows produce garbage only in rows that are never stored.
// ---------------------------------------------------------------------------
__global__ void gemm_kernel(const _Float16* __restrict__ A,
                            const _Float16* __restrict__ Wt,
                            const float* __restrict__ bias,
                            float* __restrict__ outF,
                            _Float16* __restrict__ outH,
                            const float* __restrict__ resid,
                            int rows, int Ncols, int K, int relu) {
  int wave = threadIdx.x >> 5;
  int lane = threadIdx.x & 31;
  int kh   = lane >> 4;        // half-wave: selects K chunk 0..15 / 16..31
  int lr   = lane & 15;
  int r0 = blockIdx.x * 256 + wave * 32;
  int c0 = blockIdx.y * 64;
  if (r0 >= rows) return;

  int ar0 = r0 + lr;      if (ar0 >= rows) ar0 = rows - 1;
  int ar1 = r0 + 16 + lr; if (ar1 >= rows) ar1 = rows - 1;
  const _Float16* Ap0 = A + (size_t)ar0 * K + kh * 16;
  const _Float16* Ap1 = A + (size_t)ar1 * K + kh * 16;
  const _Float16* Wp  = Wt + (size_t)(c0 + lr) * K + kh * 16;

  v8f zero = {};
  v8f acc[2][4];
#pragma unroll
  for (int i = 0; i < 2; ++i)
#pragma unroll
    for (int t = 0; t < 4; ++t) acc[i][t] = zero;

  for (int k0 = 0; k0 < K; k0 += 32) {
    v16h a0 = *(const v16h*)(Ap0 + k0);
    v16h a1 = *(const v16h*)(Ap1 + k0);
    // pull next k-step of the weight stream toward the WGP (global_prefetch_b8)
    __builtin_prefetch((const void*)(Wp + k0 + 32), 0, 0);
#pragma unroll
    for (int t = 0; t < 4; ++t) {
      v16h bf = *(const v16h*)(Wp + k0 + (size_t)(16 * t) * K);
      acc[0][t] = wmma16(a0, bf, acc[0][t]);
      acc[1][t] = wmma16(a1, bf, acc[1][t]);
    }
  }

#pragma unroll
  for (int i = 0; i < 2; ++i) {
#pragma unroll
    for (int t = 0; t < 4; ++t) {
      int col = c0 + t * 16 + lr;
      float bv = bias ? bias[col] : 0.f;
#pragma unroll
      for (int j = 0; j < 8; ++j) {
        int row = r0 + i * 16 + kh * 8 + j;
        if (row < rows) {
          size_t o = (size_t)row * Ncols + col;
          float v = acc[i][t][j] + bv;
          if (resid) v += resid[o];
          if (relu) v = fmaxf(v, 0.f);
          if (outF) outF[o] = v;
          if (outH) outH[o] = (_Float16)v;
        }
      }
    }
  }
}

// ---------------------------------------------------------------------------
// LayerNorm over last dim (=256). One wave per row; 8 elems/lane.
// ---------------------------------------------------------------------------
__global__ void ln_kernel(const float* __restrict__ x,
                          const float* __restrict__ g,
                          const float* __restrict__ b,
                          float* __restrict__ outF,
                          _Float16* __restrict__ outH, int rows) {
  int wave = threadIdx.x >> 5, lane = threadIdx.x & 31;
  int row = blockIdx.x * 8 + wave;
  if (row >= rows) return;
  float vals[8];
  float s = 0.f;
#pragma unroll
  for (int k = 0; k < 8; ++k) {
    float v = x[(size_t)row * HIDN + lane + k * 32];
    vals[k] = v; s += v;
  }
  for (int o = 1; o < 32; o <<= 1) s += __shfl_xor(s, o, 32);
  float mean = s * (1.f / HIDN);
  float vs = 0.f;
#pragma unroll
  for (int k = 0; k < 8; ++k) { float d = vals[k] - mean; vs += d * d; }
  for (int o = 1; o < 32; o <<= 1) vs += __shfl_xor(vs, o, 32);
  float rstd = rsqrtf(vs * (1.f / HIDN) + LNEPS);
#pragma unroll
  for (int k = 0; k < 8; ++k) {
    int d = lane + k * 32;
    float y = (vals[k] - mean) * rstd * g[d] + b[d];
    size_t o = (size_t)row * HIDN + d;
    if (outF) outF[o] = y;
    if (outH) outH[o] = (_Float16)y;
  }
}

// ---------------------------------------------------------------------------
// Front-end: x = LN(f2 + f3 + sinenc(c2d), ln1_g, ln1_b) -> f16
// ---------------------------------------------------------------------------
__global__ void encode_kernel(const float* __restrict__ c2d,
                              const float* __restrict__ f2,
                              const float* __restrict__ f3,
                              const float* __restrict__ g,
                              const float* __restrict__ bb,
                              _Float16* __restrict__ xh, int rows) {
  int wave = threadIdx.x >> 5, lane = threadIdx.x & 31;
  int row = blockIdx.x * 8 + wave;
  if (row >= rows) return;
  float c0 = c2d[(size_t)row * 2 + 0];
  float c1 = c2d[(size_t)row * 2 + 1];
  float vals[8];
  float s = 0.f;
#pragma unroll
  for (int k = 0; k < 8; ++k) {
    int d = lane + k * 32;
    int hh = d & 127;
    float coord = (d < 128) ? c0 : c1;
    float inv = __powf(10000.f, (float)((hh >> 1) << 1) * (1.f / 128.f));
    float ang = coord / inv;
    float pe = (hh & 1) ? __cosf(ang) : __sinf(ang);
    float v = f2[(size_t)row * CIN + d] + f3[(size_t)row * CIN + d] + pe;
    vals[k] = v; s += v;
  }
  for (int o = 1; o < 32; o <<= 1) s += __shfl_xor(s, o, 32);
  float mean = s * (1.f / CIN);
  float vs = 0.f;
#pragma unroll
  for (int k = 0; k < 8; ++k) { float d = vals[k] - mean; vs += d * d; }
  for (int o = 1; o < 32; o <<= 1) vs += __shfl_xor(vs, o, 32);
  float rstd = rsqrtf(vs * (1.f / CIN) + LNEPS);
#pragma unroll
  for (int k = 0; k < 8; ++k) {
    int d = lane + k * 32;
    xh[(size_t)row * CIN + d] = (_Float16)((vals[k] - mean) * rstd * g[d] + bb[d]);
  }
}

// ---------------------------------------------------------------------------
// Flash attention, one wave per (b, head, 16-query tile), online softmax.
// Optional query-self term (SK/SV) folded into the initial softmax state.
// Q rows are clamped (garbage rows never stored).
// ---------------------------------------------------------------------------
__global__ void attn_kernel(const _Float16* __restrict__ Qh,   // [B,Mq,256]
                            const _Float16* __restrict__ Kh,   // [B,NkPad,256]
                            const _Float16* __restrict__ Vt,   // [B,8,32,NkPad]
                            const _Float16* __restrict__ SKh,  // [B,Mq,256] | null
                            const _Float16* __restrict__ SVh,  // [B,Mq,256] | null
                            const unsigned char* __restrict__ mask, // [B,Nk] | null
                            _Float16* __restrict__ O,          // [B,Mq,256]
                            int Mq, int Nk, int NkPad) {
  int mtiles = (Mq + 15) >> 4;
  int tile = blockIdx.x;
  int b  = tile / (NHEAD * mtiles);
  int rm = tile % (NHEAD * mtiles);
  int h  = rm / mtiles;
  int mt = rm % mtiles;
  int m0 = mt * 16;
  int lane = threadIdx.x & 31;
  int kh = lane >> 4, lr = lane & 15;

  __shared__ __align__(32) _Float16 ldsP[16 * 32];
  __shared__ float ldsSelf[16];

  // Q A-fragment: row m0+lr (clamped), K chunk kh*16
  v16h qa;
  {
    int m = m0 + lr; if (m >= Mq) m = Mq - 1;
    qa = *(const v16h*)(Qh + ((size_t)(b * Mq + m)) * HIDN + h * DHH + kh * 16);
  }

  float mrow[8], lsum[8];
  v8f acc0 = {}, acc1 = {};

  if (SKh) {
    // self score per row: 2 lanes per row, 16 products each
    int r = lane >> 1, part = lane & 1;
    float s = 0.f;
    int m = r; if (m0 + m < Mq) {
      const _Float16* qp = Qh  + ((size_t)(b * Mq + m0 + m)) * HIDN + h * DHH + part * 16;
      const _Float16* kp = SKh + ((size_t)(b * Mq + m0 + m)) * HIDN + h * DHH + part * 16;
#pragma unroll
      for (int j = 0; j < 16; ++j) s += (float)qp[j] * (float)kp[j];
    }
    s += __shfl_xor(s, 1, 32);
    if (part == 0) ldsSelf[r] = s;
    __syncthreads();
#pragma unroll
    for (int j = 0; j < 8; ++j) {
      mrow[j] = ldsSelf[kh * 8 + j] * RSCALE;
      lsum[j] = 1.f;
      int m2 = m0 + kh * 8 + j;
      if (m2 < Mq) {
        acc0[j] = (float)SVh[((size_t)(b * Mq + m2)) * HIDN + h * DHH + lr];
        acc1[j] = (float)SVh[((size_t)(b * Mq + m2)) * HIDN + h * DHH + 16 + lr];
      }
    }
    __syncthreads();
  } else {
#pragma unroll
    for (int j = 0; j < 8; ++j) { mrow[j] = -3.0e38f; lsum[j] = 0.f; }
  }

  const _Float16* Kb = Kh + (size_t)b * NkPad * HIDN + h * DHH + kh * 16;
  const _Float16* Vb = Vt + ((size_t)(b * NHEAD + h)) * DHH * NkPad;

  for (int n0 = 0; n0 < NkPad; n0 += 32) {
    // scores: two 16x16 tiles (key columns n0..n0+15 and n0+16..n0+31)
    v16h kb0 = *(const v16h*)(Kb + (size_t)(n0 + lr) * HIDN);
    v16h kb1 = *(const v16h*)(Kb + (size_t)(n0 + 16 + lr) * HIDN);
    v8f z = {};
    v8f s0 = wmma16(qa, kb0, z);
    v8f s1 = wmma16(qa, kb1, z);

    int nc0 = n0 + lr, nc1 = n0 + 16 + lr;
    bool dead0 = nc0 >= Nk, dead1 = nc1 >= Nk;
    float sub0 = (!dead0 && mask && !mask[(size_t)b * Nk + nc0]) ? 1e10f : 0.f;
    float sub1 = (!dead1 && mask && !mask[(size_t)b * Nk + nc1]) ? 1e10f : 0.f;

#pragma unroll
    for (int j = 0; j < 8; ++j) {
      float a0 = dead0 ? -3.0e38f : (s0[j] - sub0) * RSCALE;
      float a1 = dead1 ? -3.0e38f : (s1[j] - sub1) * RSCALE;
      float rmax = fmaxf(a0, a1);
      for (int o = 1; o < 16; o <<= 1) rmax = fmaxf(rmax, __shfl_xor(rmax, o, 32));
      float nm = fmaxf(mrow[j], rmax);
      float sc = __expf(mrow[j] - nm);
      float p0 = dead0 ? 0.f : __expf(a0 - nm);
      float p1 = dead1 ? 0.f : __expf(a1 - nm);
      float ps = p0 + p1;
      for (int o = 1; o < 16; o <<= 1) ps += __shfl_xor(ps, o, 32);
      mrow[j] = nm;
      lsum[j] = lsum[j] * sc + ps;
      acc0[j] *= sc;
      acc1[j] *= sc;
      ldsP[(kh * 8 + j) * 32 + lr]      = (_Float16)p0;
      ldsP[(kh * 8 + j) * 32 + 16 + lr] = (_Float16)p1;
    }
    __syncthreads();
    // P as A-fragment (16x32), V^T as B-fragments
    v16h pa  = *(const v16h*)(ldsP + (size_t)lr * 32 + kh * 16);
    v16h vb0 = *(const v16h*)(Vb + (size_t)(lr)      * NkPad + n0 + kh * 16);
    v16h vb1 = *(const v16h*)(Vb + (size_t)(16 + lr) * NkPad + n0 + kh * 16);
    acc0 = wmma16(pa, vb0, acc0);
    acc1 = wmma16(pa, vb1, acc1);
    __syncthreads();
  }

#pragma unroll
  for (int j = 0; j < 8; ++j) {
    int m = m0 + kh * 8 + j;
    if (m < Mq) {
      float inv = lsum[j] > 0.f ? 1.f / lsum[j] : 0.f;
      size_t o = ((size_t)(b * Mq + m)) * HIDN + h * DHH;
      O[o + lr]      = (_Float16)(acc0[j] * inv);
      O[o + 16 + lr] = (_Float16)(acc1[j] * inv);
    }
  }
}

// ---------------------------------------------------------------------------
// Small glue kernels
// ---------------------------------------------------------------------------
__global__ void wconv_kernel(const float* __restrict__ in, _Float16* __restrict__ out,
                             int K, int N) {  // in [K,N] f32 -> out [N,K] f16
  size_t idx = (size_t)blockIdx.x * blockDim.x + threadIdx.x;
  if (idx >= (size_t)K * N) return;
  size_t n = idx / K, k = idx % K;
  out[idx] = (_Float16)in[k * N + n];
}

__global__ void vtrans_kernel(const _Float16* __restrict__ V, _Float16* __restrict__ Vt,
                              int NkPad) {  // V [B,NkPad,256] -> Vt [B,8,32,NkPad]
  size_t idx = (size_t)blockIdx.x * blockDim.x + threadIdx.x;
  size_t total = (size_t)BB * NHEAD * DHH * NkPad;
  if (idx >= total) return;
  size_t n = idx % NkPad, r = idx / NkPad;
  int d = r % DHH; r /= DHH;
  int h = r % NHEAD; int b = r / NHEAD;
  Vt[idx] = V[((size_t)b * NkPad + n) * HIDN + h * DHH + d];
}

__global__ void concat_kernel(const _Float16* __restrict__ cls_h,
                              const _Float16* __restrict__ ks_h,
                              const _Float16* __restrict__ qs_h,
                              _Float16* __restrict__ kcat) {
  size_t idx = (size_t)blockIdx.x * blockDim.x + threadIdx.x;
  if (idx >= (size_t)BB * NCATP * HIDN) return;
  int d = idx % HIDN;
  size_t r = idx / HIDN;
  int row = r % NCATP;
  int b = r / NCATP;
  _Float16 v = (_Float16)0.f;
  if (row == 0)            v = cls_h[(size_t)b * HIDN + d];
  else if (row <= NN)      v = ks_h[((size_t)b * NN + (row - 1)) * HIDN + d];
  else if (row <= NN + MM) v = qs_h[((size_t)b * MM + (row - 1 - NN)) * HIDN + d];
  kcat[idx] = v;
}

__global__ void clsinit_kernel(const float* __restrict__ cls_f3d,
                               float* __restrict__ cf, _Float16* __restrict__ ch) {
  int idx = blockIdx.x * blockDim.x + threadIdx.x;
  if (idx >= BB * HIDN) return;
  float v = cls_f3d[idx % HIDN];
  cf[idx] = v;
  ch[idx] = (_Float16)v;
}

// ---------------------------------------------------------------------------
// Host orchestration
// ---------------------------------------------------------------------------
static inline size_t alignup(size_t x) { return (x + 255) & ~(size_t)255; }

extern "C" void kernel_launch(void* const* d_in, const int* in_sizes, int n_in,
                              void* d_out, int out_size, void* d_ws, size_t ws_size,
                              hipStream_t stream) {
  (void)in_sizes; (void)n_in; (void)out_size; (void)ws_size;
  const float* qc2  = (const float*)d_in[0];
  const float* qf2  = (const float*)d_in[1];
  const float* qf3  = (const float*)d_in[2];
  const float* kc2  = (const float*)d_in[3];
  const float* kf2  = (const float*)d_in[4];
  const float* kf3  = (const float*)d_in[5];
  const unsigned char* amask = (const unsigned char*)d_in[6];
  const float* ln1g = (const float*)d_in[7];
  const float* ln1b = (const float*)d_in[8];
  const float* dcW  = (const float*)d_in[9];
  const float* dcb  = (const float*)d_in[10];
  const float* dclg = (const float*)d_in[11];
  const float* dclb = (const float*)d_in[12];
  const float* clsv = (const float*)d_in[13];
  const float* Wq = (const float*)d_in[14]; const float* bq = (const float*)d_in[15];
  const float* Wk = (const float*)d_in[16]; const float* bk = (const float*)d_in[17];
  const float* Wv = (const float*)d_in[18]; const float* bv = (const float*)d_in[19];
  const float* Wd = (const float*)d_in[20]; const float* bd = (const float*)d_in[21];
  const float* alg = (const float*)d_in[22]; const float* alb = (const float*)d_in[23];
  const float* W1 = (const float*)d_in[24]; const float* b1 = (const float*)d_in[25];
  const float* W2 = (const float*)d_in[26]; const float* b2 = (const float*)d_in[27];
  const float* flg = (const float*)d_in[28]; const float* flb = (const float*)d_in[29];

  // ---- workspace bump allocator ----
  char* w = (char*)d_ws;
  auto alloc = [&](size_t bytes) -> void* { void* p = w; w += alignup(bytes); return p; };
  const size_t QROWS = (size_t)BB * MM, KROWS = (size_t)BB * NN;
  const size_t CATROWS = (size_t)BB * NCATP;

  float*    qs_f  = (float*)alloc(QROWS * HIDN * 4);
  float*    ks_f  = (float*)alloc(KROWS * HIDN * 4);
  float*    cls_f = (float*)alloc((size_t)BB * HIDN * 4);
  _Float16* qs_h  = (_Float16*)alloc(QROWS * HIDN * 2);
  _Float16* ks_h  = (_Float16*)alloc(KROWS * HIDN * 2);
  _Float16* cls_h = (_Float16*)alloc((size_t)BB * HIDN * 2);
  _Float16* xq_h  = (_Float16*)alloc(QROWS * CIN * 2);
  _Float16* xk_h  = (_Float16*)alloc(KROWS * CIN * 2);
  _Float16* kcat  = (_Float16*)alloc(CATROWS * HIDN * 2);
  _Float16* Qbuf  = (_Float16*)alloc(KROWS * HIDN * 2);
  _Float16* Kbuf  = (_Float16*)alloc(CATROWS * HIDN * 2);
  _Float16* Vbuf  = (_Float16*)alloc(CATROWS * HIDN * 2);
  _Float16* SKbuf = (_Float16*)alloc(QROWS * HIDN * 2);
  _Float16* SVbuf = (_Float16*)alloc(QROWS * HIDN * 2);
  _Float16* Vt    = (_Float16*)alloc(CATROWS * HIDN * 2);   // [B,8,32,NkPad]
  _Float16* Obuf  = (_Float16*)alloc(KROWS * HIDN * 2);
  float*    pre_f = (float*)alloc(KROWS * HIDN * 4);
  float*    a_f   = (float*)alloc(KROWS * HIDN * 4);
  _Float16* a_h   = (_Float16*)alloc(KROWS * HIDN * 2);
  _Float16* ffn_h = (_Float16*)alloc(KROWS * INTERN * 2);
  _Float16* dcWt  = (_Float16*)alloc((size_t)CIN * HIDN * 2);
  _Float16* WqT = (_Float16*)alloc((size_t)NLAY * HIDN * HIDN * 2);
  _Float16* WkT = (_Float16*)alloc((size_t)NLAY * HIDN * HIDN * 2);
  _Float16* WvT = (_Float16*)alloc((size_t)NLAY * HIDN * HIDN * 2);
  _Float16* WdT = (_Float16*)alloc((size_t)NLAY * HIDN * HIDN * 2);
  _Float16* W1T = (_Float16*)alloc((size_t)NLAY * HIDN * INTERN * 2);
  _Float16* W2T = (_Float16*)alloc((size_t)NLAY * HIDN * INTERN * 2);

  auto gemm = [&](const _Float16* A, const _Float16* Wt, const float* bias,
                  float* oF, _Float16* oH, const float* resid,
                  int rows, int Ncols, int K, int relu) {
    dim3 grid((rows + 255) / 256, Ncols / 64);
    gemm_kernel<<<grid, 256, 0, stream>>>(A, Wt, bias, oF, oH, resid, rows, Ncols, K, relu);
  };
  auto ln = [&](const float* x, const float* g, const float* b, float* oF,
                _Float16* oH, int rows) {
    ln_kernel<<<(rows + 7) / 8, 256, 0, stream>>>(x, g, b, oF, oH, rows);
  };
  auto wconv = [&](const float* in, _Float16* out, int K, int N) {
    size_t tot = (size_t)K * N;
    wconv_kernel<<<(unsigned)((tot + 255) / 256), 256, 0, stream>>>(in, out, K, N);
  };

  // ---- weight prep: f32 [K,N] -> f16 [N,K] ----
  wconv(dcW, dcWt, CIN, HIDN);
  for (int i = 0; i < NLAY; ++i) {
    wconv(Wq + (size_t)i * HIDN * HIDN, WqT + (size_t)i * HIDN * HIDN, HIDN, HIDN);
    wconv(Wk + (size_t)i * HIDN * HIDN, WkT + (size_t)i * HIDN * HIDN, HIDN, HIDN);
    wconv(Wv + (size_t)i * HIDN * HIDN, WvT + (size_t)i * HIDN * HIDN, HIDN, HIDN);
    wconv(Wd + (size_t)i * HIDN * HIDN, WdT + (size_t)i * HIDN * HIDN, HIDN, HIDN);
    wconv(W1 + (size_t)i * HIDN * INTERN, W1T + (size_t)i * HIDN * INTERN, HIDN, INTERN);
    wconv(W2 + (size_t)i * HIDN * INTERN, W2T + (size_t)i * HIDN * INTERN, INTERN, HIDN);
  }

  // ---- front-end: sinenc + LN1 + dc GEMM + dcLN ----
  encode_kernel<<<(unsigned)((QROWS + 7) / 8), 256, 0, stream>>>(qc2, qf2, qf3, ln1g, ln1b, xq_h, (int)QROWS);
  encode_kernel<<<(unsigned)((KROWS + 7) / 8), 256, 0, stream>>>(kc2, kf2, kf3, ln1g, ln1b, xk_h, (int)KROWS);
  gemm(xq_h, dcWt, dcb, pre_f, nullptr, nullptr, (int)QROWS, HIDN, CIN, 0);
  ln(pre_f, dclg, dclb, qs_f, qs_h, (int)QROWS);
  gemm(xk_h, dcWt, dcb, pre_f, nullptr, nullptr, (int)KROWS, HIDN, CIN, 0);
  ln(pre_f, dclg, dclb, ks_f, ks_h, (int)KROWS);
  clsinit_kernel<<<2, 256, 0, stream>>>(clsv, cls_f, cls_h);

  // ---- transformer layers ----
  auto run_layer = [&](int i, float* q_f, _Float16* q_h, int Mq,
                       const _Float16* kin_h, int NkPad, int Nk,
                       const unsigned char* maskp, bool self) {
    int rowsQ = BB * Mq, rowsK = BB * NkPad;
    const _Float16* wq = WqT + (size_t)i * HIDN * HIDN;
    const _Float16* wk = WkT + (size_t)i * HIDN * HIDN;
    const _Float16* wv = WvT + (size_t)i * HIDN * HIDN;
    const _Float16* wd = WdT + (size_t)i * HIDN * HIDN;
    gemm(q_h,  wq, bq + (size_t)i * HIDN, nullptr, Qbuf, nullptr, rowsQ, HIDN, HIDN, 0);
    gemm(kin_h, wk, bk + (size_t)i * HIDN, nullptr, Kbuf, nullptr, rowsK, HIDN, HIDN, 0);
    gemm(kin_h, wv, bv + (size_t)i * HIDN, nullptr, Vbuf, nullptr, rowsK, HIDN, HIDN, 0);
    if (self) {
      gemm(q_h, wk, bk + (size_t)i * HIDN, nullptr, SKbuf, nullptr, rowsQ, HIDN, HIDN, 0);
      gemm(q_h, wv, bv + (size_t)i * HIDN, nullptr, SVbuf, nullptr, rowsQ, HIDN, HIDN, 0);
    }
    size_t vtn = (size_t)BB * NHEAD * DHH * NkPad;
    vtrans_kernel<<<(unsigned)((vtn + 255) / 256), 256, 0, stream>>>(Vbuf, Vt, NkPad);
    int mtiles = (Mq + 15) / 16;
    attn_kernel<<<BB * NHEAD * mtiles, 32, 0, stream>>>(
        Qbuf, Kbuf, Vt, self ? SKbuf : nullptr, self ? SVbuf : nullptr,
        maskp, Obuf, Mq, Nk, NkPad);
    gemm(Obuf, wd, bd + (size_t)i * HIDN, pre_f, nullptr, q_f, rowsQ, HIDN, HIDN, 0);
    ln(pre_f, alg + (size_t)i * HIDN, alb + (size_t)i * HIDN, a_f, a_h, rowsQ);
    gemm(a_h, W1T + (size_t)i * HIDN * INTERN, b1 + (size_t)i * INTERN,
         nullptr, ffn_h, nullptr, rowsQ, INTERN, HIDN, 1);
    gemm(ffn_h, W2T + (size_t)i * HIDN * INTERN, b2 + (size_t)i * HIDN,
         pre_f, nullptr, a_f, rowsQ, HIDN, INTERN, 0);
    ln(pre_f, flg + (size_t)i * HIDN, flb + (size_t)i * HIDN, q_f, q_h, rowsQ);
  };

  for (int i = 0; i < NLAY; ++i) {
    run_layer(i, ks_f, ks_h, NN, ks_h, NN, NN, amask, false);          // ks self-attn
    run_layer(i, qs_f, qs_h, MM, ks_h, NN, NN, amask, true);           // qs cross + self
    size_t cn = (size_t)BB * NCATP * HIDN;
    concat_kernel<<<(unsigned)((cn + 255) / 256), 256, 0, stream>>>(cls_h, ks_h, qs_h, kcat);
    run_layer(i, cls_f, cls_h, 1, kcat, NCATP, NCAT, nullptr, false);  // cls cross-attn
  }

  // ---- outputs: (qs, ks, cls) concatenated flat, f32 ----
  float* out = (float*)d_out;
  hipMemcpyAsync(out, qs_f, QROWS * HIDN * 4, hipMemcpyDeviceToDevice, stream);
  hipMemcpyAsync(out + QROWS * HIDN, ks_f, KROWS * HIDN * 4, hipMemcpyDeviceToDevice, stream);
  hipMemcpyAsync(out + QROWS * HIDN + KROWS * HIDN, cls_f,
                 (size_t)BB * HIDN * 4, hipMemcpyDeviceToDevice, stream);
}
